// XlAttention_57561151701368
// MI455X (gfx1250) — matrix operand order repaired and checked by reading, not compiled, not run on hardware
//
#include <hip/hip_runtime.h>

#define DEV __device__ __forceinline__

typedef __attribute__((ext_vector_type(16))) __bf16 v16bf;
typedef __attribute__((ext_vector_type(8)))  __bf16 v8bf;
typedef __attribute__((ext_vector_type(8)))  float  v8f;
typedef __attribute__((ext_vector_type(4)))  unsigned int u32x4;
typedef int v4i_async __attribute__((__vector_size__(4 * sizeof(int))));

static constexpr int B_  = 4;
static constexpr int NH  = 8;
static constexpr int QL  = 1024;
static constexpr int ML  = 1024;
static constexpr int KL  = 2048;   // QL + ML
static constexpr int DM  = 512;
static constexpr int DH  = 64;
static constexpr int RROWS = 2176; // KL + 128 zero pad (block-shared j range)
static constexpr float SCALE = 0.125f; // 1/sqrt(DH)
static constexpr float NEG_BIG = -3.0e38f;

// ---- CDNA5 async global->LDS path (ASYNCcnt), with safe fallbacks ----
#if defined(__gfx1250__) && __has_builtin(__builtin_amdgcn_global_load_async_to_lds_b128)
#define HAVE_ASYNC 1
#else
#define HAVE_ASYNC 0
#endif

#if HAVE_ASYNC
#if __has_builtin(__builtin_amdgcn_s_wait_asynccnt)
#define WAIT_ASYNC(n) __builtin_amdgcn_s_wait_asynccnt(n)
#else
#define WAIT_ASYNC(n) asm volatile("s_wait_asynccnt %0" ::"i"(n))
#endif
#else
#define WAIT_ASYNC(n)
#endif

union Frag {
  v16bf v;
  v8bf  h[2];
  u32x4 q[2];
};

DEV v8f wmma_bf16(const Frag& a, const Frag& b, v8f c) {
  return __builtin_amdgcn_wmma_f32_16x16x32_bf16(false, a.v, false, b.v,
                                                 (short)0, c, false, false);
}

// A-operand (16x32 bf16): lane L -> row M = L%16.
// lanes 0-15 hold K 0..7 & 16..23 ; lanes 16-31 hold K 8..15 & 24..31.
DEV Frag loadA(const __bf16* base, int row0, int row_stride, int k0, int lane) {
  int r = row0 + (lane & 15);
  int kh = k0 + ((lane >> 4) << 3);
  const __bf16* p = base + (long)r * row_stride + kh;
  Frag f;
  f.q[0] = *(const u32x4*)(p);
  f.q[1] = *(const u32x4*)(p + 16);
  return f;
}

// B-operand (32x16 bf16) loaded from B^T stored row-major [N][k_stride]:
// lane L -> col N = n0 + L%16 ; lanes 0-15 hold K k0+0..7 & k0+16..23, lanes 16-31 K k0+8..15 & k0+24..31.
DEV Frag loadB(const __bf16* baseT, int n0, int k_stride, int k0, int lane) {
  int c = n0 + (lane & 15);
  int k = k0 + ((lane >> 4) << 3);
  const __bf16* p = baseT + (long)c * k_stride + k;
  Frag f;
  f.q[0] = *(const u32x4*)(p);
  f.q[1] = *(const u32x4*)(p + 16);
  return f;
}

// ---------------- conversions ----------------
__global__ __launch_bounds__(256) void cvt_bf16_kernel(const float* __restrict__ src,
                                                       __bf16* __restrict__ dst, int n) {
  int i = blockIdx.x * blockDim.x + threadIdx.x;
  int stride = gridDim.x * blockDim.x;
  for (; i < n; i += stride) dst[i] = (__bf16)src[i];
}

// W:[DM][NH][DH] (f32) -> Wt:[NH][DH][DM] (bf16)  (B^T layout for projections)
__global__ __launch_bounds__(256) void cvt_weight_kernel(const float* __restrict__ W,
                                                         __bf16* __restrict__ Wt) {
  int i = blockIdx.x * blockDim.x + threadIdx.x; // NH*DH*DM threads exactly
  int k = i % DM;
  int a = (i / DM) % DH;
  int n = i / (DM * DH);
  Wt[i] = (__bf16)W[(k * NH + n) * DH + a];
}

// ---------------- projection core: 16 rows x 64 cols, K = 512 ----------------
DEV void proj_tile16x64(const __bf16* __restrict__ Arows, int row0,
                        const __bf16* __restrict__ WtN, int lane, v8f acc[4]) {
  for (int kk = 0; kk < DM / 32; ++kk) {
    Frag a = loadA(Arows, row0, DM, kk * 32, lane);
#pragma unroll
    for (int at = 0; at < 4; ++at) {
      Frag b = loadB(WtN, at * 16, DM, kk * 32, lane);
      acc[at] = wmma_bf16(a, b, acc[at]);
    }
  }
}

__global__ __launch_bounds__(256)
void proj_q_kernel(const __bf16* __restrict__ xb, const __bf16* __restrict__ Wqt,
                   const float* __restrict__ bias_k, const float* __restrict__ bias_r,
                   __bf16* __restrict__ QK, __bf16* __restrict__ QR) {
  int lane = threadIdx.x & 31;
  int gid  = blockIdx.x * (blockDim.x >> 5) + (threadIdx.x >> 5);
  int tile = gid & 63;       // QL/16
  int bn   = gid >> 6;
  int n = bn & (NH - 1), b = bn >> 3;
  int l0 = tile * 16;
  v8f acc[4] = {};
  proj_tile16x64(xb, b * KL + ML + l0, Wqt + (long)n * DH * DM, lane, acc);
  long obase = ((long)bn * QL + l0) * DH;
  int col = lane & 15, hf = lane >> 4;
#pragma unroll
  for (int at = 0; at < 4; ++at) {
    float bk = bias_k[n * DH + at * 16 + col];
    float br = bias_r[n * DH + at * 16 + col];
#pragma unroll
    for (int t = 0; t < 8; ++t) {
      long o = obase + (long)(t + 8 * hf) * DH + at * 16 + col;
      QK[o] = (__bf16)(acc[at][t] + bk);
      QR[o] = (__bf16)(acc[at][t] + br);
    }
  }
}

__global__ __launch_bounds__(256)
void proj_k_kernel(const __bf16* __restrict__ xb, const __bf16* __restrict__ Wkt,
                   __bf16* __restrict__ Kb) {
  int lane = threadIdx.x & 31;
  int gid  = blockIdx.x * (blockDim.x >> 5) + (threadIdx.x >> 5);
  int tile = gid & 127;      // KL/16
  int bn   = gid >> 7;
  int n = bn & (NH - 1), b = bn >> 3;
  int l0 = tile * 16;
  v8f acc[4] = {};
  proj_tile16x64(xb, b * KL + l0, Wkt + (long)n * DH * DM, lane, acc);
  long obase = ((long)bn * KL + l0) * DH;
  int col = lane & 15, hf = lane >> 4;
#pragma unroll
  for (int at = 0; at < 4; ++at)
#pragma unroll
    for (int t = 0; t < 8; ++t)
      Kb[obase + (long)(t + 8 * hf) * DH + at * 16 + col] = (__bf16)acc[at][t];
}

// V stored transposed: VT [bn][DH][KL]
__global__ __launch_bounds__(256)
void proj_v_kernel(const __bf16* __restrict__ xb, const __bf16* __restrict__ Wvt,
                   __bf16* __restrict__ VT) {
  int lane = threadIdx.x & 31;
  int gid  = blockIdx.x * (blockDim.x >> 5) + (threadIdx.x >> 5);
  int tile = gid & 127;
  int bn   = gid >> 7;
  int n = bn & (NH - 1), b = bn >> 3;
  int l0 = tile * 16;
  v8f acc[4] = {};
  proj_tile16x64(xb, b * KL + l0, Wvt + (long)n * DH * DM, lane, acc);
  long obase = (long)bn * DH * KL;
  int col = lane & 15, hf = lane >> 4;
#pragma unroll
  for (int at = 0; at < 4; ++at)
#pragma unroll
    for (int t = 0; t < 8; ++t)
      VT[obase + (long)(at * 16 + col) * KL + l0 + t + 8 * hf] = (__bf16)acc[at][t];
}

// R: [NH][RROWS][DH], rows >= KL zeroed (pad for shifted windows)
__global__ __launch_bounds__(256)
void proj_r_kernel(const __bf16* __restrict__ pb, const __bf16* __restrict__ Wpt,
                   __bf16* __restrict__ Rb) {
  int lane = threadIdx.x & 31;
  int gid  = blockIdx.x * (blockDim.x >> 5) + (threadIdx.x >> 5);
  const int TPN = RROWS / 16; // 136
  int tile = gid % TPN;
  int n    = gid / TPN;
  int l0 = tile * 16;
  long obase = ((long)n * RROWS + l0) * DH;
  int col = lane & 15, hf = lane >> 4;
  if (l0 >= KL) {
#pragma unroll
    for (int at = 0; at < 4; ++at)
#pragma unroll
      for (int t = 0; t < 8; ++t)
        Rb[obase + (long)(t + 8 * hf) * DH + at * 16 + col] = (__bf16)0.0f;
    return;
  }
  v8f acc[4] = {};
  proj_tile16x64(pb, l0, Wpt + (long)n * DH * DM, lane, acc);
#pragma unroll
  for (int at = 0; at < 4; ++at)
#pragma unroll
    for (int t = 0; t < 8; ++t)
      Rb[obase + (long)(t + 8 * hf) * DH + at * 16 + col] = (__bf16)acc[at][t];
}

// ---------------- fused flash attention (block-shared async K/V staging) ----------------
#define WPB 8
__global__ __launch_bounds__(256)
void attn_kernel(const __bf16* __restrict__ QK, const __bf16* __restrict__ QR,
                 const __bf16* __restrict__ Kb, const __bf16* __restrict__ VT,
                 const __bf16* __restrict__ Rb, float* __restrict__ O) {
  __shared__ __align__(16) __bf16 Ks[2][32 * DH]; // K chunk: 32 rows x 64
  __shared__ __align__(16) __bf16 Vs[2][DH * 32]; // V^T chunk: 64 rows x 32
  __shared__ float bds[WPB][48][18];              // 3 BD windows/wave, padded rows

  int tid  = threadIdx.x;
  int lane = tid & 31;
  int wid  = tid >> 5;
  int bn = blockIdx.x >> 3;       // one (b,n) per block
  int ib = blockIdx.x & 7;        // 128-row i-band
  int n  = bn & (NH - 1);
  int i0 = ib * 128 + wid * 16;   // this wave's i-tile
  const __bf16* qk = QK + (long)bn * QL * DH;
  const __bf16* qr = QR + (long)bn * QL * DH;
  const __bf16* kb = Kb + (long)bn * KL * DH;
  const __bf16* vt = VT + (long)bn * DH * KL;
  const __bf16* rb = Rb + (long)n * RROWS * DH;

  // staging addresses for this thread (one B128 each for K and V^T per chunk)
  const __bf16* gk0 = kb + (long)(tid >> 3) * DH + (tid & 7) * 8;   // + j0*DH
  const __bf16* gv0 = vt + (long)(tid >> 2) * KL + (tid & 3) * 8;   // + j0
  __bf16* lk = &Ks[0][(tid >> 3) * DH + (tid & 7) * 8];
  __bf16* lv = &Vs[0][(tid >> 2) * 32 + (tid & 3) * 8];
  const int lbufK = 32 * DH, lbufV = DH * 32;

#if HAVE_ASYNC
#define STAGE_ONE(g, l)                                                       \
  __builtin_amdgcn_global_load_async_to_lds_b128((v4i_async*)(void*)(g),      \
                                                 (v4i_async*)(void*)(l), 0, 0)
#else
#define STAGE_ONE(g, l) *(u32x4*)(l) = *(const u32x4*)(g)
#endif
#define STAGE(bufi, j0s)                                                      \
  do {                                                                        \
    const __bf16* gk_ = gk0 + (long)(j0s) * DH;                               \
    const __bf16* gv_ = gv0 + (j0s);                                          \
    __bf16* lk_ = lk + (bufi) * lbufK;                                        \
    __bf16* lv_ = lv + (bufi) * lbufV;                                        \
    STAGE_ONE(gk_, lk_);                                                      \
    STAGE_ONE(gv_, lv_);                                                      \
  } while (0)

  // B fragments of Q (constant over the j loop): S^T = K_tile x Q^T
  Frag bqk[2], bqr[2];
#pragma unroll
  for (int kk = 0; kk < 2; ++kk) {
    bqk[kk] = loadB(qk, i0, DH, kk * 32, lane);
    bqr[kk] = loadB(qr, i0, DH, kk * 32, lane);
  }

  int il = lane & 15, hf = lane >> 4;
  float m = NEG_BIG, lsum = 0.0f;
  v8f o[4] = {};

  const int nch = ib * 4 + 36; // covers j <= ib*128 + 127 + ML
  STAGE(0, 0);
  for (int c = 0; c < nch; ++c) {
    int j0 = c * 32;
    int buf = c & 1;
    if (c + 1 < nch) {
      STAGE(buf ^ 1, j0 + 32);
      WAIT_ASYNC(2);
    } else {
      WAIT_ASYNC(0);
    }
    __syncthreads(); // chunk c staged by all waves

    // --- AC: two S^T tiles (j rows, i cols), contraction over DH=64 ---
    v8f st[2] = {};
#pragma unroll
    for (int s = 0; s < 2; ++s)
#pragma unroll
      for (int kk = 0; kk < 2; ++kk) {
        Frag ak = loadA(&Ks[buf][0], 16 * s, DH, kk * 32, lane);
        st[s] = wmma_bf16(ak, bqk[kk], st[s]);
      }
    // --- BD: three 16-row r-windows starting at c0 = j0 - i0 + QL - 16 ---
    int c0 = j0 - i0 + (QL - 16);
#pragma unroll
    for (int w = 0; w < 3; ++w) {
      v8f wt = {};
#pragma unroll
      for (int kk = 0; kk < 2; ++kk) {
        Frag ar = loadA(rb, c0 + 16 * w, DH, kk * 32, lane);
        wt = wmma_bf16(ar, bqr[kk], wt);
      }
#pragma unroll
      for (int t = 0; t < 8; ++t)
        bds[wid][w * 16 + t + 8 * hf][il] = wt[t];
    }
    // --- diagonal gather: BD_shift[j,i] = window[(j - i) + 15][i]; add, scale, mask ---
#pragma unroll
    for (int s = 0; s < 2; ++s)
#pragma unroll
      for (int t = 0; t < 8; ++t) {
        int jl = 16 * s + t + 8 * hf;
        float bd = bds[wid][jl - il + 15][il];
        float sv = SCALE * (st[s][t] + bd);
        st[s][t] = ((j0 + jl) > (i0 + il + ML)) ? NEG_BIG : sv;
      }
    // --- online softmax (row = i = lane%16, halves combined via shfl_xor 16) ---
    float rm = NEG_BIG;
#pragma unroll
    for (int s = 0; s < 2; ++s)
#pragma unroll
      for (int t = 0; t < 8; ++t) rm = fmaxf(rm, st[s][t]);
    rm = fmaxf(rm, __shfl_xor(rm, 16, 32));
    float newm = fmaxf(m, rm);
    float alpha = __expf(m - newm);
    m = newm;
    float rs = 0.0f;
    Frag ap; // P in A-operand layout: lane-local repack from S^T C-layout
#pragma unroll
    for (int s = 0; s < 2; ++s)
#pragma unroll
      for (int t = 0; t < 8; ++t) {
        float pv = __expf(st[s][t] - newm);
        rs += pv;
        ap.h[s][t] = (__bf16)pv;
      }
    rs += __shfl_xor(rs, 16, 32);
    lsum = lsum * alpha + rs;
    // rescale O accumulators (row of O tile = t + 8*half -> alpha at that lane)
#pragma unroll
    for (int t = 0; t < 8; ++t) {
      float at_ = __shfl(alpha, t + 8 * hf, 32);
#pragma unroll
      for (int dt = 0; dt < 4; ++dt) o[dt][t] *= at_;
    }
    // --- P·V: O[i, d-tile] += P(16x32) x V^T-frag(32x16) ---
#pragma unroll
    for (int dt = 0; dt < 4; ++dt) {
      Frag bv = loadB(&Vs[buf][0], dt * 16, 32, 0, lane);
      o[dt] = wmma_bf16(ap, bv, o[dt]);
    }
    __syncthreads(); // all waves done with chunk c before its buffer is re-staged
  }
  // normalize and store O (f32, [bn][QL][DH])
  float rinv = 1.0f / lsum;
  float* Op = O + ((long)bn * QL + i0) * DH;
#pragma unroll
  for (int t = 0; t < 8; ++t) {
    float rv = __shfl(rinv, t + 8 * hf, 32);
#pragma unroll
    for (int dt = 0; dt < 4; ++dt)
      Op[(long)(t + 8 * hf) * DH + dt * 16 + il] = o[dt][t] * rv;
  }
#undef STAGE
#undef STAGE_ONE
}

// ---------------- output projection: out[b,l,dm] = sum_k O[b,l,k] * Wout[dm,k] ----------------
__global__ __launch_bounds__(256)
void outproj_kernel(const float* __restrict__ O, const float* __restrict__ Wout,
                    float* __restrict__ out) {
  int lane = threadIdx.x & 31;
  int gid  = blockIdx.x * (blockDim.x >> 5) + (threadIdx.x >> 5);
  int dmt = gid & 31;   // DM/16 tiles
  int rt  = gid >> 5;   // (B*QL)/16 tiles
  int l0g = rt * 16;
  int b = l0g >> 10, l0 = l0g & (QL - 1);
  int dm0 = dmt * 16;
  v8f acc = {};
  for (int kk = 0; kk < 16; ++kk) {
    int k0 = kk * 32;
    Frag a;
    {
      int r   = l0 + (lane & 15);
      int kb1 = k0 + ((lane >> 4) << 3);
      int kb2 = kb1 + 16;
      const float* p1 = O + (((long)(b * NH + kb1 / DH) * QL + r) * DH + (kb1 % DH));
      const float* p2 = O + (((long)(b * NH + kb2 / DH) * QL + r) * DH + (kb2 % DH));
#pragma unroll
      for (int e = 0; e < 8; ++e) {
        a.h[0][e] = (__bf16)p1[e];
        a.h[1][e] = (__bf16)p2[e];
      }
    }
    Frag bf;
    {
      int dm = dm0 + (lane & 15);
      int k  = k0 + ((lane >> 4) << 3);
      const float* p = Wout + (long)dm * (NH * DH) + k;
#pragma unroll
      for (int e = 0; e < 16; ++e) bf.v[e] = (__bf16)p[e + (e >> 3) * 8];
    }
    acc = wmma_bf16(a, bf, acc);
  }
  int il = lane & 15, hf = lane >> 4;
#pragma unroll
  for (int t = 0; t < 8; ++t)
    out[((long)l0g + t + 8 * hf) * DM + dm0 + il] = acc[t];
}

// ---------------- launcher ----------------
extern "C" void kernel_launch(void* const* d_in, const int* in_sizes, int n_in,
                              void* d_out, int out_size, void* d_ws, size_t ws_size,
                              hipStream_t stream) {
  const float* x  = (const float*)d_in[0];
  const float* p  = (const float*)d_in[1];
  const float* Wq = (const float*)d_in[2];
  const float* Wk = (const float*)d_in[3];
  const float* Wv = (const float*)d_in[4];
  const float* Wo = (const float*)d_in[5];
  const float* Wp = (const float*)d_in[6];
  const float* bk = (const float*)d_in[7];
  const float* br = (const float*)d_in[8];
  // d_in[9] = mask (applied analytically), d_in[10] = qlen (compile-time)
  float* out = (float*)d_out;

  char* ws = (char*)d_ws;
  size_t off = 0;
  auto alloc = [&](size_t bytes) -> void* {
    void* q = (void*)(ws + off);
    off += (bytes + 255) & ~(size_t)255;
    return q;
  };
  __bf16* xb  = (__bf16*)alloc((size_t)B_ * KL * DM * 2);
  __bf16* pb  = (__bf16*)alloc((size_t)KL * DM * 2);
  __bf16* wqt = (__bf16*)alloc((size_t)NH * DH * DM * 2);
  __bf16* wkt = (__bf16*)alloc((size_t)NH * DH * DM * 2);
  __bf16* wvt = (__bf16*)alloc((size_t)NH * DH * DM * 2);
  __bf16* wpt = (__bf16*)alloc((size_t)NH * DH * DM * 2);
  __bf16* QK  = (__bf16*)alloc((size_t)B_ * NH * QL * DH * 2);
  __bf16* QR  = (__bf16*)alloc((size_t)B_ * NH * QL * DH * 2);
  __bf16* Kb  = (__bf16*)alloc((size_t)B_ * NH * KL * DH * 2);
  __bf16* VT  = (__bf16*)alloc((size_t)B_ * NH * DH * KL * 2);
  __bf16* Rb  = (__bf16*)alloc((size_t)NH * RROWS * DH * 2);
  float*  Of  = (float*) alloc((size_t)B_ * NH * QL * DH * 4);

  // 1. precision conversion
  cvt_bf16_kernel<<<4096, 256, 0, stream>>>(x, xb, B_ * KL * DM);
  cvt_bf16_kernel<<<1024, 256, 0, stream>>>(p, pb, KL * DM);
  cvt_weight_kernel<<<(NH * DH * DM) / 256, 256, 0, stream>>>(Wq, wqt);
  cvt_weight_kernel<<<(NH * DH * DM) / 256, 256, 0, stream>>>(Wk, wkt);
  cvt_weight_kernel<<<(NH * DH * DM) / 256, 256, 0, stream>>>(Wv, wvt);
  cvt_weight_kernel<<<(NH * DH * DM) / 256, 256, 0, stream>>>(Wp, wpt);

  // 2. projections (WMMA)
  proj_q_kernel<<<(B_ * NH * (QL / 16)) / 8, 256, 0, stream>>>(xb, wqt, bk, br, QK, QR);
  proj_k_kernel<<<(B_ * NH * (KL / 16)) / 8, 256, 0, stream>>>(xb, wkt, Kb);
  proj_v_kernel<<<(B_ * NH * (KL / 16)) / 8, 256, 0, stream>>>(xb, wvt, VT);
  proj_r_kernel<<<(NH * (RROWS / 16)) / 8, 256, 0, stream>>>(pb, wpt, Rb);

  // 3. fused flash attention (one (b,n) x 128-row band per block, async K/V staging)
  attn_kernel<<<B_ * NH * 8, 256, 0, stream>>>(QK, QR, Kb, VT, Rb, Of);

  // 4. output projection (WMMA)
  outproj_kernel<<<((B_ * QL / 16) * (DM / 16)) / 8, 256, 0, stream>>>(Of, Wo, out);

  (void)in_sizes; (void)n_in; (void)out_size; (void)ws_size;
}
